// LSTMCell_317827580479
// MI455X (gfx1250) — compile-verified, matrix operand order checked by reading, not run on hardware
//
#include <hip/hip_runtime.h>

// ---------------------------------------------------------------------------
// LSTM cell fused as 4x bf16-WMMA GEMM + gate epilogue for gfx1250 (MI455X).
// out = (h_new[32768x512], c_new[32768x512]) fp32, concatenated.
// ws layout: [0, 64MB)   : xc bf16  [32768][1024]  (concat(h, x))
//            [64MB,68MB) : W  bf16  [4][512][1024] (gate order f,i,g,o)
// Weight K-chunks DMA'd to double-buffered LDS one chunk ahead via
// GLOBAL_LOAD_ASYNC_TO_LDS_B128 (ASYNCcnt, in-order completion).
// K loop fully unrolled; DMA source pointer is a single loop-carried VGPR
// pair ("+v" asm operand) to avoid pre-materializing 31 absolute addresses.
// ---------------------------------------------------------------------------

typedef unsigned short u16;
typedef unsigned int   u32;
typedef __attribute__((ext_vector_type(4)))  u32    u32x4;
typedef __attribute__((ext_vector_type(4)))  float  f32x4;
typedef __attribute__((ext_vector_type(16))) __bf16 v16bf;
typedef __attribute__((ext_vector_type(8)))  float  v8f;

union BFrag { u32x4 q[2]; v16bf v; };

__device__ __forceinline__ u32 f2bf_pack(float lo, float hi) {
  // fp32 -> bf16 round-to-nearest-even, packed pair
  u32 a = __float_as_uint(lo);
  u32 b = __float_as_uint(hi);
  a = (a + 0x7FFFu + ((a >> 16) & 1u)) >> 16;
  b = (b + 0x7FFFu + ((b >> 16) & 1u)) >> 16;
  return (a & 0xFFFFu) | (b << 16);
}

__device__ __forceinline__ float fsigmoid(float v) {
  return 1.0f / (1.0f + __expf(-v));
}
__device__ __forceinline__ float ftanh(float v) {
  return 2.0f / (1.0f + __expf(-2.0f * v)) - 1.0f;
}

// 32B per-lane async DMA: global -> LDS, no VGPR staging, tracked by ASYNCcnt.
// IOFFSET applies to both global and LDS addresses (CDNA5 ISA),
// so the global K-offset cannot go into IOFFSET; instead the source pointer is
// a read-write operand advanced afterwards, keeping ONE loop-carried register
// pair instead of 31 hoisted absolute addresses.
__device__ __forceinline__ void async_copy32(u32 ldsoff, const u16*& gp) {
  asm volatile(
      "global_load_async_to_lds_b128 %1, %0, off\n\t"
      "global_load_async_to_lds_b128 %1, %0, off offset:16"
      : "+v"(gp)
      : "v"(ldsoff)
      : "memory");
  gp += 32;  // advance one K-chunk (64 bytes)
}

// --- Prepass 1: xc = concat(h, x) as bf16, row-major [B][1024] ---------------
__global__ __launch_bounds__(256) void convert_xc(
    const float* __restrict__ h, const float* __restrict__ x,
    u16* __restrict__ xcB) {
  const int id = blockIdx.x * 256 + threadIdx.x;   // 32768*128 chunks of 8
  const int b  = id >> 7;
  const int k0 = (id & 127) << 3;
  const float* src = (k0 < 512) ? (h + b * 512 + k0)
                                : (x + b * 512 + (k0 - 512));
  f32x4 lo = *(const f32x4*)(src);
  f32x4 hi = *(const f32x4*)(src + 4);
  u32x4 o;
  o.x = f2bf_pack(lo.x, lo.y);
  o.y = f2bf_pack(lo.z, lo.w);
  o.z = f2bf_pack(hi.x, hi.y);
  o.w = f2bf_pack(hi.z, hi.w);
  *(u32x4*)(xcB + (size_t)b * 1024 + k0) = o;
}

// --- Prepass 2: weights -> bf16, gate-major [4][512][1024] -------------------
__global__ __launch_bounds__(256) void convert_w(
    const float* __restrict__ Wf, const float* __restrict__ Wi,
    const float* __restrict__ Wg, const float* __restrict__ Wo,
    u16* __restrict__ Wb) {
  const int g = blockIdx.y;
  const float* W = (g == 0) ? Wf : (g == 1) ? Wi : (g == 2) ? Wg : Wo;
  const int id  = blockIdx.x * 256 + threadIdx.x;  // 65536 chunks of 8
  const int off = id << 3;
  f32x4 lo = *(const f32x4*)(W + off);
  f32x4 hi = *(const f32x4*)(W + off + 4);
  u32x4 o;
  o.x = f2bf_pack(lo.x, lo.y);
  o.y = f2bf_pack(lo.z, lo.w);
  o.z = f2bf_pack(hi.x, hi.y);
  o.w = f2bf_pack(hi.z, hi.w);
  *(u32x4*)(Wb + (size_t)g * 524288 + off) = o;
}

// --- Main: fused 4-gate WMMA GEMM + LSTM epilogue ----------------------------
// Block tile: 128 M x 32 N x 4 gates; 8 waves, wave w owns rows [16w,16w+16).
// Each wave: 2 N-subtiles x 4 gates accumulators (v8f each).
__global__ __launch_bounds__(256) void lstm_gemm_wmma(
    const u16* __restrict__ xcB, const u16* __restrict__ Wb,
    const float* __restrict__ bfp, const float* __restrict__ bip,
    const float* __restrict__ bgp, const float* __restrict__ bop,
    const float* __restrict__ cin, float* __restrict__ out) {
  __shared__ u16 lB[2][4 * 32 * 32];  // 2 x 8KB: [buf][gate][n<32][k<32]

  const int tid  = threadIdx.x;
  const int wave = tid >> 5;
  const int lane = tid & 31;
  const int lm   = lane & 15;   // N (B-frag) / M (A-frag) index within subtile
  const int grp  = lane >> 4;   // K-half selector per CDNA5 ISA layout
  const int nbase = blockIdx.y * 32;

  // Cooperative B staging: 128 rows (4 gates x 32 n), 2 halves of 32B each.
  const int brow  = tid >> 1;
  const int bgate = brow >> 5;
  const int bn    = brow & 31;
  const int bhalf = tid & 1;
  const u16* bptr = Wb + (size_t)bgate * (512 * 1024) +
                    (size_t)(nbase + bn) * 1024 + bhalf * 16;

  // LDS byte addresses for the async DMA's VDST operand. Deriving them from
  // the real array pointer (ptrtoint) yields the addrspace(3) byte offset
  // (low 32 bits of the LDS-aperture flat address) AND escapes lB so alias
  // analysis keeps the ds_load reads of the asm-written data alive.
  const int slot = bgate * 1024 + bn * 32 + bhalf * 16;
  const u32 lds0 = (u32)(size_t)(&lB[0][slot]);
  const u32 lds1 = (u32)(size_t)(&lB[1][slot]);

  const int mrow  = blockIdx.x * 128 + wave * 16 + lm;
  const u16* arow = xcB + (size_t)mrow * 1024 + grp * 8;

  v8f acc[2][4];
  const v8f vzero = {0.f, 0.f, 0.f, 0.f, 0.f, 0.f, 0.f, 0.f};
#pragma unroll
  for (int nt = 0; nt < 2; ++nt)
#pragma unroll
    for (int gi = 0; gi < 4; ++gi) acc[nt][gi] = vzero;

  // Prologue: DMA chunk 0 into buffer 0 (bptr advances to chunk 1).
  async_copy32(lds0, bptr);

#pragma unroll
  for (int it = 0; it < 32; ++it) {
    const int kk = it * 32;

    // DMA chunk it+1 one iteration ahead into the other buffer. Safe: every
    // wave executed s_wait_dscnt 0 + barrier after last reading that buffer.
    if (it < 31) async_copy32((it & 1) ? lds0 : lds1, bptr);

    // A fragment from global while the DMAs are in flight.
    BFrag a;
    a.q[0] = *(const u32x4*)(arow + kk);        // K = kk + grp*8 + [0,8)
    a.q[1] = *(const u32x4*)(arow + kk + 16);   // K = kk + 16 + grp*8 + [0,8)
    __builtin_prefetch(arow + kk + 64, 0, 3);   // A two chunks ahead

    // Async loads complete in order: counter <= 2 means chunk `it`'s two
    // transfers have landed while chunk it+1's two may still be in flight.
    if (it < 31)
      asm volatile("s_wait_asynccnt 2" ::: "memory");
    else
      asm volatile("s_wait_asynccnt 0" ::: "memory");
    __syncthreads();  // chunk `it` visible to all waves

    // Load ALL 8 B fragments into distinct registers so the ds_loads can
    // pipeline, then issue 8 back-to-back WMMAs into the XDL pipe.
    const u16* base = &lB[it & 1][0];
    BFrag bfr[4][2];
#pragma unroll
    for (int gi = 0; gi < 4; ++gi) {
      const u16* pb = base + gi * 1024 + grp * 8;
      bfr[gi][0].q[0] = *(const u32x4*)(pb + lm * 32);
      bfr[gi][0].q[1] = *(const u32x4*)(pb + lm * 32 + 16);
      bfr[gi][1].q[0] = *(const u32x4*)(pb + (lm + 16) * 32);
      bfr[gi][1].q[1] = *(const u32x4*)(pb + (lm + 16) * 32 + 16);
    }
#pragma unroll
    for (int gi = 0; gi < 4; ++gi) {
      acc[0][gi] = __builtin_amdgcn_wmma_f32_16x16x32_bf16(
          false, a.v, false, bfr[gi][0].v, (short)0, acc[0][gi], false, false);
      acc[1][gi] = __builtin_amdgcn_wmma_f32_16x16x32_bf16(
          false, a.v, false, bfr[gi][1].v, (short)0, acc[1][gi], false, false);
    }

    // All LDS reads of this buffer must be COMPLETE (DScnt=0), not just
    // issued, before any wave is allowed to overwrite it next iteration.
    asm volatile("s_wait_dscnt 0" ::: "memory");
    __syncthreads();
  }

  // Epilogue: C/D layout -> M = grp*8 + vgpr, N = lm (per CDNA5 ISA 7.12.2)
  const int mtop = blockIdx.x * 128 + wave * 16 + grp * 8;
#pragma unroll
  for (int nt = 0; nt < 2; ++nt) {
    const int col = nbase + nt * 16 + lm;
    const float bfv = bfp[col], biv = bip[col];
    const float bgv = bgp[col], bov = bop[col];
#pragma unroll
    for (int r = 0; r < 8; ++r) {
      const int row = mtop + r;
      const float fg = fsigmoid(acc[nt][0][r] + bfv);
      const float ig = fsigmoid(acc[nt][1][r] + biv);
      const float gg = ftanh   (acc[nt][2][r] + bgv);
      const float og = fsigmoid(acc[nt][3][r] + bov);
      const float cold = cin[row * 512 + col];
      const float cn   = cold * fg + ig * gg;
      const float hn   = og * ftanh(cn);
      out[row * 512 + col]               = hn;   // h_new
      out[32768 * 512 + row * 512 + col] = cn;   // c_new
    }
  }
}

// ---------------------------------------------------------------------------
extern "C" void kernel_launch(void* const* d_in, const int* in_sizes, int n_in,
                              void* d_out, int out_size, void* d_ws,
                              size_t ws_size, hipStream_t stream) {
  (void)in_sizes; (void)n_in; (void)out_size; (void)ws_size;
  const float* x  = (const float*)d_in[0];
  const float* h  = (const float*)d_in[1];
  const float* c  = (const float*)d_in[2];
  const float* Wf = (const float*)d_in[3];
  const float* bf = (const float*)d_in[4];
  const float* Wi = (const float*)d_in[5];
  const float* bi = (const float*)d_in[6];
  const float* Wo = (const float*)d_in[7];
  const float* bo = (const float*)d_in[8];
  const float* Wg = (const float*)d_in[9];
  const float* bg = (const float*)d_in[10];
  float* out = (float*)d_out;

  u16* xcB = (u16*)d_ws;                                     // 64 MiB
  u16* Wb  = (u16*)((char*)d_ws + (size_t)32768 * 1024 * 2); // +4 MiB

  convert_xc<<<16384, 256, 0, stream>>>(h, x, xcB);
  convert_w<<<dim3(256, 4), 256, 0, stream>>>(Wf, Wi, Wg, Wo, Wb);
  lstm_gemm_wmma<<<dim3(256, 16), 256, 0, stream>>>(
      xcB, Wb, bf, bi, bg, bo, c, out);
}
